// GCN_7241314861628
// MI455X (gfx1250) — matrix operand order repaired and checked by reading, not compiled, not run on hardware
//
#include <hip/hip_runtime.h>
#include <math.h>

typedef __attribute__((ext_vector_type(16))) _Float16 v16h;
typedef __attribute__((ext_vector_type(8)))  float    v8f;

#define NN   50000
#define EDGES 150000
#define HH   32
#define CLS  13
#define KCHUNKS 11      // 10 outer-product chunks (k fixed per chunk) + 1 raw-hsrc (bias) chunk
#define KTOT 352

// ---------------------------------------------------------------- zero fill
__global__ __launch_bounds__(256) void zero_kernel(float* __restrict__ p, int n) {
  int i = blockIdx.x * 256 + threadIdx.x;
  if (i < n) p[i] = 0.f;
}

// ---------------------------------------------------------------- edge messages
// m = Z @ W2aug ; Z[e, k*32+i] = ee[e,k]*hsrc[e,i], Z[e,320+i] = hsrc[e,i]
// W2aug[k*32+i, o] = W2[k, i*32+o] ; W2aug[320+i, o] = b2[i*32+o]
__global__ __launch_bounds__(256) void edge_msg_kernel(
    const float* __restrict__ h,          // [N,32]
    const int*   __restrict__ srcIdx,     // [E]
    const int*   __restrict__ dstIdx,     // [E]
    const float* __restrict__ ea,         // [E,2]
    const float* __restrict__ w1,         // [2,10]
    const float* __restrict__ bias1,      // [10]
    const float* __restrict__ w2,         // [10,1024]
    const float* __restrict__ bias2,      // [1024]
    float* __restrict__ agg)              // [N,32] (atomic accum)
{
  __shared__ _Float16 Bt[32][360];        // B transposed: Bt[col][K], padded stride
  __shared__ _Float16 Asrc[8][16][40];    // per-wave: 16 edges x 32 src features (f16)
  __shared__ float    Ee[8][16][12];      // per-wave: 16 edges x 11 chunk scalars

  // cooperative stage of the layer-constant 352x32 B matrix
  for (int idx = threadIdx.x; idx < KTOT * 32; idx += 256) {
    int f = idx >> 5, n = idx & 31;
    float v;
    if (f < 320) { int k = f >> 5, i = f & 31; v = w2[k * 1024 + i * 32 + n]; }
    else         { int i = f - 320;            v = bias2[i * 32 + n]; }
    Bt[n][f] = (_Float16)v;
  }
  __syncthreads();

  const int wave = threadIdx.x >> 5, lane = threadIdx.x & 31;
  const int tile = blockIdx.x * 8 + wave;          // one 16-edge tile per wave
  if (tile * 16 >= EDGES) return;
  const int e0 = tile * 16;

  // stage edge-MLP outputs (lanes 0..15, one edge each)
  if (lane < 16) {
    int e = e0 + lane;
    if (e < EDGES) {
      float a0 = ea[2 * e], a1 = ea[2 * e + 1];
      #pragma unroll
      for (int j = 0; j < 10; j++) {
        float v = fmaf(a0, w1[j], fmaf(a1, w1[10 + j], bias1[j]));
        Ee[wave][lane][j] = v > 0.f ? v : 0.f;
      }
      Ee[wave][lane][10] = 1.f;                    // bias/raw-hsrc chunk scale
    } else {
      #pragma unroll
      for (int j = 0; j < 11; j++) Ee[wave][lane][j] = 0.f;
    }
  }

  // stage gathered h[src] rows as f16 (2 lanes per edge row)
  {
    int m = lane >> 1, ch = (lane & 1) * 16;
    int e = e0 + m;
    if (e < EDGES) {
      const float4* hp = (const float4*)(h + (size_t)srcIdx[e] * HH + ch);
      #pragma unroll
      for (int q = 0; q < 4; q++) {
        float4 v = hp[q];
        Asrc[wave][m][ch + 4 * q + 0] = (_Float16)v.x;
        Asrc[wave][m][ch + 4 * q + 1] = (_Float16)v.y;
        Asrc[wave][m][ch + 4 * q + 2] = (_Float16)v.z;
        Asrc[wave][m][ch + 4 * q + 3] = (_Float16)v.w;
      }
    } else {
      #pragma unroll
      for (int q = 0; q < 16; q++) Asrc[wave][m][ch + q] = (_Float16)0.f;
    }
  }

  // WMMA: A 16x32 f16, contiguous k order chosen here (B matches) -> lanes 0-15: K 0-15 ; lanes 16-31: K 16-31
  const int row  = lane & 15;
  const int koff = (lane < 16) ? 0 : 16;
  const _Float16* ap = &Asrc[wave][row][koff];

  v8f c0 = {}, c1 = {};
  #pragma unroll
  for (int kc = 0; kc < KCHUNKS; kc++) {
    _Float16 es = (_Float16)Ee[wave][row][kc];
    v16h a;
    #pragma unroll
    for (int t = 0; t < 8; t++) { a[t] = ap[t] * es; a[8 + t] = ap[8 + t] * es; }
    int bk = kc * 32 + ((lane < 16) ? 0 : 16);     // B 32x16: lanes 0-15 K 0-15, 16-31 K 16-31
    const _Float16* bp0 = &Bt[row][bk];            // output cols 0..15
    const _Float16* bp1 = &Bt[row + 16][bk];       // output cols 16..31
    v16h bf0, bf1;
    #pragma unroll
    for (int t = 0; t < 16; t++) { bf0[t] = bp0[t]; bf1[t] = bp1[t]; }
    c0 = __builtin_amdgcn_wmma_f32_16x16x32_f16(false, a, false, bf0, (short)0, c0, false, false);
    c1 = __builtin_amdgcn_wmma_f32_16x16x32_f16(false, a, false, bf1, (short)0, c1, false, false);
  }

  // D layout: VGPR r -> (M = r or 8+r, N = lane&15); scatter-add to agg[dst]
  const int ncol = lane & 15;
  const int mb = (lane < 16) ? 0 : 8;
  #pragma unroll
  for (int r = 0; r < 8; r++) {
    int e = e0 + mb + r;
    if (e < EDGES) {
      size_t d = (size_t)dstIdx[e] * HH;
      atomicAdd(&agg[d + ncol], c0[r]);
      atomicAdd(&agg[d + ncol + 16], c1[r]);
    }
  }
}

// ---------------------------------------------------------------- node update + BN stats
// hpre = agg + h @ root + cbias ; stats[c]=sum, stats[32+c]=sumsq
__global__ __launch_bounds__(256) void node_update_kernel(
    const float* __restrict__ h, const float* __restrict__ agg,
    const float* __restrict__ rootW,      // [32,32]
    const float* __restrict__ cbias,      // [32]
    float* __restrict__ hpre, float* __restrict__ stats)
{
  __shared__ _Float16 Rt[32][36];
  __shared__ _Float16 Ah[8][16][40];

  for (int idx = threadIdx.x; idx < 32 * 32; idx += 256) {
    int k = idx >> 5, n = idx & 31;
    Rt[n][k] = (_Float16)rootW[k * 32 + n];
  }
  __syncthreads();

  const int wave = threadIdx.x >> 5, lane = threadIdx.x & 31;
  const int tile = blockIdx.x * 8 + wave;
  if (tile * 16 >= NN) return;
  const int n0 = tile * 16;

  {
    int m = lane >> 1, ch = (lane & 1) * 16;
    int node = n0 + m;
    if (node < NN) {
      const float4* hp = (const float4*)(h + (size_t)node * HH + ch);
      #pragma unroll
      for (int q = 0; q < 4; q++) {
        float4 v = hp[q];
        Ah[wave][m][ch + 4 * q + 0] = (_Float16)v.x;
        Ah[wave][m][ch + 4 * q + 1] = (_Float16)v.y;
        Ah[wave][m][ch + 4 * q + 2] = (_Float16)v.z;
        Ah[wave][m][ch + 4 * q + 3] = (_Float16)v.w;
      }
    } else {
      #pragma unroll
      for (int q = 0; q < 16; q++) Ah[wave][m][ch + q] = (_Float16)0.f;
    }
  }

  const int row  = lane & 15;
  const int koff = (lane < 16) ? 0 : 16;
  const _Float16* ap = &Ah[wave][row][koff];
  v16h a;
  #pragma unroll
  for (int t = 0; t < 8; t++) { a[t] = ap[t]; a[8 + t] = ap[8 + t]; }

  int bk = (lane < 16) ? 0 : 16;
  const _Float16* bp0 = &Rt[row][bk];
  const _Float16* bp1 = &Rt[row + 16][bk];
  v16h bf0, bf1;
  #pragma unroll
  for (int t = 0; t < 16; t++) { bf0[t] = bp0[t]; bf1[t] = bp1[t]; }

  const int ncol = lane & 15;
  const int mb = (lane < 16) ? 0 : 8;
  v8f c0, c1;
  #pragma unroll
  for (int r = 0; r < 8; r++) {              // preload C with agg + conv bias
    int node = n0 + mb + r;
    bool ok = node < NN;
    c0[r] = ok ? agg[(size_t)node * HH + ncol] + cbias[ncol] : 0.f;
    c1[r] = ok ? agg[(size_t)node * HH + ncol + 16] + cbias[ncol + 16] : 0.f;
  }
  c0 = __builtin_amdgcn_wmma_f32_16x16x32_f16(false, a, false, bf0, (short)0, c0, false, false);
  c1 = __builtin_amdgcn_wmma_f32_16x16x32_f16(false, a, false, bf1, (short)0, c1, false, false);

  float s0 = 0.f, q0 = 0.f, s1 = 0.f, q1 = 0.f;
  #pragma unroll
  for (int r = 0; r < 8; r++) {
    int node = n0 + mb + r;
    if (node < NN) {
      hpre[(size_t)node * HH + ncol] = c0[r];
      hpre[(size_t)node * HH + ncol + 16] = c1[r];
      s0 += c0[r]; q0 += c0[r] * c0[r];
      s1 += c1[r]; q1 += c1[r] * c1[r];
    }
  }
  atomicAdd(&stats[ncol], s0);
  atomicAdd(&stats[32 + ncol], q0);
  atomicAdd(&stats[ncol + 16], s1);
  atomicAdd(&stats[32 + ncol + 16], q1);
}

// ---------------------------------------------------------------- BN apply + leaky relu (in place)
__global__ __launch_bounds__(256) void bn_apply_kernel(
    float* __restrict__ hp, const float* __restrict__ stats,
    const float* __restrict__ gamma, const float* __restrict__ beta)
{
  int i = blockIdx.x * 256 + threadIdx.x;
  if (i >= NN * HH) return;
  int c = i & 31;
  float mu  = stats[c] * (1.f / NN);
  float var = stats[32 + c] * (1.f / NN) - mu * mu;
  float v = (hp[i] - mu) * rsqrtf(var + 1e-5f) * gamma[c] + beta[c];
  hp[i] = v > 0.f ? v : 0.01f * v;
}

// ---------------------------------------------------------------- FC head + log_softmax
__global__ __launch_bounds__(256) void head_kernel(
    const float* __restrict__ h, const float* __restrict__ fcW,
    const float* __restrict__ fcb, float* __restrict__ out)
{
  int node = blockIdx.x * 256 + threadIdx.x;
  if (node >= NN) return;
  float hr[32];
  const float4* hp = (const float4*)(h + (size_t)node * HH);
  #pragma unroll
  for (int q = 0; q < 8; q++) {
    float4 v = hp[q];
    hr[4 * q] = v.x; hr[4 * q + 1] = v.y; hr[4 * q + 2] = v.z; hr[4 * q + 3] = v.w;
  }
  float lg[CLS]; float mx = -1e30f;
  #pragma unroll
  for (int c = 0; c < CLS; c++) {
    float s = fcb[c];
    #pragma unroll
    for (int i = 0; i < 32; i++) s = fmaf(hr[i], fcW[i * CLS + c], s);
    lg[c] = s; mx = fmaxf(mx, s);
  }
  float se = 0.f;
  #pragma unroll
  for (int c = 0; c < CLS; c++) se += __expf(lg[c] - mx);
  float lse = mx + __logf(se);
  #pragma unroll
  for (int c = 0; c < CLS; c++) out[(size_t)node * CLS + c] = lg[c] - lse;
}

// ---------------------------------------------------------------- launcher
extern "C" void kernel_launch(void* const* d_in, const int* in_sizes, int n_in,
                              void* d_out, int out_size, void* d_ws, size_t ws_size,
                              hipStream_t stream) {
  (void)in_sizes; (void)n_in; (void)out_size; (void)ws_size;
  const float* x    = (const float*)d_in[0];
  const int*   ei   = (const int*)d_in[1];
  const float* ea   = (const float*)d_in[2];
  const float* nW1  = (const float*)d_in[3];
  const float* nb1  = (const float*)d_in[4];
  const float* nW2  = (const float*)d_in[5];
  const float* nb2  = (const float*)d_in[6];
  const float* rt   = (const float*)d_in[7];
  const float* cb   = (const float*)d_in[8];
  const float* gm   = (const float*)d_in[9];
  const float* bt   = (const float*)d_in[10];
  const float* fcW  = (const float*)d_in[11];
  const float* fcb  = (const float*)d_in[12];
  const int* srcIdx = ei;
  const int* dstIdx = ei + EDGES;

  float* hA    = (float*)d_ws;
  float* hB    = hA + (size_t)NN * HH;
  float* agg   = hB + (size_t)NN * HH;
  float* stats = agg + (size_t)NN * HH;   // 64 floats, contiguous after agg

  const int edgeBlocks = ((EDGES / 16) + 7) / 8;    // 1172 (8 waves/block, 16 edges/wave)
  const int nodeBlocks = (((NN + 15) / 16) + 7) / 8;

  const float* hcur = x;
  for (int i = 0; i < 3; i++) {
    float* hnext = (i & 1) ? hB : hA;
    zero_kernel<<<(NN * HH + 64 + 255) / 256, 256, 0, stream>>>(agg, NN * HH + 64);
    edge_msg_kernel<<<edgeBlocks, 256, 0, stream>>>(hcur, srcIdx, dstIdx, ea,
        nW1 + i * 20, nb1 + i * 10, nW2 + i * 10240, nb2 + i * 1024, agg);
    node_update_kernel<<<nodeBlocks, 256, 0, stream>>>(hcur, agg, rt + i * 1024,
        cb + i * 32, hnext, stats);
    bn_apply_kernel<<<(NN * HH + 255) / 256, 256, 0, stream>>>(hnext, stats,
        gm + i * 32, bt + i * 32);
    hcur = hnext;
  }
  head_kernel<<<(NN + 255) / 256, 256, 0, stream>>>(hcur, fcW, fcb, (float*)d_out);
}
